// MeshNetworkPEARAR_86303072845943
// MI455X (gfx1250) — matrix operand order repaired and checked by reading, not compiled, not run on hardware
//
#include <hip/hip_runtime.h>
#include <hip/hip_bf16.h>
#include <math.h>

#define P_    10000
#define NP_   50
#define EP_   200
#define IN_   64
#define HIDE_ 256
#define C2_   64
#define RD_   128
#define HIDM_ 256
#define OUT_  16
#define EM_   160000
#define EPS_  1e-5f
#define SLOPE_ 0.01f

typedef __attribute__((ext_vector_type(16))) _Float16 v16h;
typedef __attribute__((ext_vector_type(8)))  float    v8f;

__device__ __forceinline__ float lrelu(float x) { return x >= 0.f ? x : SLOPE_ * x; }

// ---------------------------------------------------------------------------
// Async global->LDS copy (CDNA5 GLOBAL_LOAD_ASYNC_TO_LDS_B128, ASYNCcnt).
// VDST lane value = LDS byte offset; VADDR = 64-bit global address.
// Generic pointers to __shared__ hold the LDS offset in their low 32 bits.
// ---------------------------------------------------------------------------
__device__ __forceinline__ void async_copy16(void* lds_dst, const void* gsrc) {
    unsigned lds_off = (unsigned)(size_t)lds_dst;
    asm volatile("global_load_async_to_lds_b128 %0, %1, off"
                 :: "v"(lds_off), "v"(gsrc) : "memory");
}
__device__ __forceinline__ void async_wait0() {
    asm volatile("s_wait_asynccnt 0" ::: "memory");
}

// K offset for 16-bit A-matrix 16x32 fragment (ISA 7.12.2 layout)
__device__ __forceinline__ int a_koff(int i, int half) {
    return ((i < 4) ? (2 * i) : (16 + 2 * (i - 4))) + 8 * half;
}

// A fragment (16x32, f16) from row-major f32 source
__device__ __forceinline__ v16h load_a_f32(const float* src, int lda, int row_base, int k_base, int lane) {
    int row = row_base + (lane & 15);
    int half = lane >> 4;
    v16h a;
#pragma unroll
    for (int i = 0; i < 8; ++i) {
        int k = k_base + a_koff(i, half);
        a[2 * i]     = (_Float16)src[row * lda + k];
        a[2 * i + 1] = (_Float16)src[row * lda + k + 1];
    }
    return a;
}

// A fragment with per-row scale folded in (for rsqrt(deg_in) normalization)
__device__ __forceinline__ v16h load_a_f32_scaled(const float* src, int lda, int row_base, int k_base,
                                                  int lane, const float* rowscale) {
    int row = row_base + (lane & 15);
    int half = lane >> 4;
    float s = rowscale[row];
    v16h a;
#pragma unroll
    for (int i = 0; i < 8; ++i) {
        int k = k_base + a_koff(i, half);
        a[2 * i]     = (_Float16)(src[row * lda + k] * s);
        a[2 * i + 1] = (_Float16)(src[row * lda + k + 1] * s);
    }
    return a;
}

// B fragment (32x16, f16) from row-major f32 weight
__device__ __forceinline__ v16h load_b_f32(const float* w, int ldb, int k_base, int col_base, int lane) {
    int col = col_base + (lane & 15);
    int half = lane >> 4;
    v16h b;
#pragma unroll
    for (int i = 0; i < 8; ++i) {
        int k = k_base + a_koff(i, half);
        b[2 * i]     = (_Float16)w[k * ldb + col];
        b[2 * i + 1] = (_Float16)w[(k + 1) * ldb + col];
    }
    return b;
}

__device__ __forceinline__ v8f wmma_f16(v16h a, v16h b, v8f c) {
    return __builtin_amdgcn_wmma_f32_16x16x32_f16(false, a, false, b, (short)0, c, false, false);
}

// Unconditional D-tile stores (targets padded to full 16-row tiles -> no exec churn)
__device__ __forceinline__ void store_d_f32(float* dst, int ldd, int row_base, int col_base,
                                            int lane, v8f d) {
    int half = lane >> 4;
    int col = col_base + (lane & 15);
#pragma unroll
    for (int v = 0; v < 8; ++v)
        dst[(row_base + half * 8 + v) * ldd + col] = d[v];
}

__device__ __forceinline__ void store_d_f16(_Float16* dst, int ldd, int row_base, int col_base,
                                            int lane, v8f d) {
    int half = lane >> 4;
    int col = col_base + (lane & 15);
#pragma unroll
    for (int v = 0; v < 8; ++v)
        dst[(row_base + half * 8 + v) * ldd + col] = (_Float16)d[v];
}

// ---------------------------------------------------------------------------
// Kernel 1: per-patch embedder. One block (4 waves) per patch.
// ---------------------------------------------------------------------------
__global__ __launch_bounds__(128)
void patch_embed_kernel(const float* __restrict__ feats,
                        const int* __restrict__ srcs, const int* __restrict__ dsts,
                        const float* __restrict__ ews,
                        const float* __restrict__ W1, const float* __restrict__ g1, const float* __restrict__ b1,
                        const float* __restrict__ W2, const float* __restrict__ g2, const float* __restrict__ b2,
                        const float* __restrict__ We,
                        float* __restrict__ emb_out) {
    const int p = blockIdx.x;
    const int tid = threadIdx.x;
    const int lane = tid & 31;
    const int wave = tid >> 5;

    __shared__ __align__(16) float    sF[NP_ * 64];   // feat [50,64]
    __shared__ __align__(16) float    sA[64 * 64];    // conv aggregation (rows 50..63 stay 0)
    __shared__ __align__(16) _Float16 sH[64 * 256];   // hidden h1 (padded to 64 rows); first half reused as f32 GEMM2 out
    __shared__ float sDegO[64], sDegI[64];
    __shared__ float sR[512];                          // [r0(64)|r1(256)|r2(64)|emb(128)]

    const float* feat = feats + (size_t)p * NP_ * IN_;
    const int*   src  = srcs + (size_t)p * EP_;
    const int*   dst  = dsts + (size_t)p * EP_;
    const float* ew   = ews  + (size_t)p * EP_;

    // async DMA of the 12.8KB feat tile into LDS (800 x 16B)
    for (int i = tid; i < (NP_ * 64) / 4; i += 128)
        async_copy16(&sF[i * 4], &feat[i * 4]);
    for (int i = tid; i < 64 * 64; i += 128) sA[i] = 0.f;
    if (tid < 64) { sDegO[tid] = 0.f; sDegI[tid] = 0.f; }
    async_wait0();
    __syncthreads();

    // unweighted degrees
    for (int e = tid; e < EP_; e += 128) {
        atomicAdd(&sDegO[src[e]], 1.f);
        atomicAdd(&sDegI[dst[e]], 1.f);
    }
    __syncthreads();
    if (tid < 64) {
        sDegO[tid] = rsqrtf(fmaxf(sDegO[tid], 1.f));
        sDegI[tid] = rsqrtf(fmaxf(sDegI[tid], 1.f));
        // r0 = feat.mean(0)
        float s = 0.f;
        for (int r = 0; r < NP_; ++r) s += sF[r * 64 + tid];
        sR[tid] = s * (1.f / NP_);
    }
    __syncthreads();

    // conv1 scatter: agg[dst] += feat[src] * rsqrt(degO[src]) * ew
    for (int idx = tid; idx < EP_ * 64; idx += 128) {
        int e = idx >> 6, c = idx & 63;
        int s = src[e];
        atomicAdd(&sA[dst[e] * 64 + c], sF[s * 64 + c] * sDegO[s] * ew[e]);
    }
    __syncthreads();
    for (int i = tid; i < NP_ * 64; i += 128) sA[i] *= sDegI[i >> 6];
    __syncthreads();

    // GEMM1: [64x64] @ W1[64x256] -> sH (f16), wave = M-tile
    {
        v16h a0 = load_a_f32(sA, 64, wave * 16, 0,  lane);
        v16h a1 = load_a_f32(sA, 64, wave * 16, 32, lane);
        for (int nt = 0; nt < 16; ++nt) {
            v8f acc = {};
            v16h b0 = load_b_f32(W1, 256, 0,  nt * 16, lane);
            v16h b1v = load_b_f32(W1, 256, 32, nt * 16, lane);
            acc = wmma_f16(a0, b0, acc);
            acc = wmma_f16(a1, b1v, acc);
            store_d_f16(sH, 256, wave * 16, nt * 16, lane, acc);
        }
    }
    __syncthreads();

    // graph_norm1 + lrelu + r1 (each thread owns 2 channels; rows < 50 only)
    for (int cc = 0; cc < 2; ++cc) {
        int c = tid + cc * 128;
        float s = 0.f, q = 0.f;
        for (int r = 0; r < NP_; ++r) { float v = (float)sH[r * 256 + c]; s += v; q += v * v; }
        float mu = s * (1.f / NP_);
        float var = q * (1.f / NP_) - mu * mu;
        float rs = rsqrtf(var + EPS_);
        float rsum = 0.f;
        for (int r = 0; r < NP_; ++r) {
            float v = lrelu(g1[c] * ((float)sH[r * 256 + c] - mu) * rs + b1[c]);
            sH[r * 256 + c] = (_Float16)v;
            rsum += v;
        }
        sR[64 + c] = rsum * (1.f / NP_);
    }
    __syncthreads();

    // conv2 + GEMM2, K-chunked by 64 channels; accumulators live across chunks
    v8f z8 = {};
    v8f acc2[4] = { z8, z8, z8, z8 };
    for (int ch = 0; ch < 4; ++ch) {
        const int cb = ch * 64;
        for (int i = tid; i < 64 * 64; i += 128) sA[i] = 0.f;
        __syncthreads();
        for (int idx = tid; idx < EP_ * 64; idx += 128) {
            int e = idx >> 6, c = idx & 63;
            int s = src[e];
            atomicAdd(&sA[dst[e] * 64 + c], (float)sH[s * 256 + cb + c] * sDegO[s] * ew[e]);
        }
        __syncthreads();
        for (int i = tid; i < NP_ * 64; i += 128) sA[i] *= sDegI[i >> 6];
        __syncthreads();
        v16h a0 = load_a_f32(sA, 64, wave * 16, 0,  lane);
        v16h a1 = load_a_f32(sA, 64, wave * 16, 32, lane);
#pragma unroll
        for (int nt = 0; nt < 4; ++nt) {
            v16h b0 = load_b_f32(W2, 64, cb + 0,  nt * 16, lane);
            v16h b1v = load_b_f32(W2, 64, cb + 32, nt * 16, lane);
            acc2[nt] = wmma_f16(a0, b0, acc2[nt]);
            acc2[nt] = wmma_f16(a1, b1v, acc2[nt]);
        }
        __syncthreads();   // protect sA / last sH reads before next chunk's zero
    }
    // sH is dead now; reuse its storage as the f32 [64x64] GEMM2 output
    float* sOut = (float*)sH;
#pragma unroll
    for (int nt = 0; nt < 4; ++nt) store_d_f32(sOut, 64, wave * 16, nt * 16, lane, acc2[nt]);
    __syncthreads();

    // graph_norm2 + lrelu + r2 (only the readout mean is needed downstream)
    if (tid < 64) {
        int c = tid;
        float s = 0.f, q = 0.f;
        for (int r = 0; r < NP_; ++r) { float v = sOut[r * 64 + c]; s += v; q += v * v; }
        float mu = s * (1.f / NP_);
        float var = q * (1.f / NP_) - mu * mu;
        float rs = rsqrtf(var + EPS_);
        float rsum = 0.f;
        for (int r = 0; r < NP_; ++r)
            rsum += lrelu(g2[c] * (sOut[r * 64 + c] - mu) * rs + b2[c]);
        sR[320 + c] = rsum * (1.f / NP_);
    }
    __syncthreads();

    // embed: [384] @ We[384x128]; one output column per thread
    {
        float acc = 0.f;
        for (int k = 0; k < 384; ++k) acc += sR[k] * We[k * RD_ + tid];
        sR[384 + tid] = acc;
    }
    __syncthreads();
    // per-row InstanceNorm (no affine) + lrelu
    if (tid == 0) {
        float s = 0.f, q = 0.f;
        for (int i = 0; i < RD_; ++i) { float v = sR[384 + i]; s += v; q += v * v; }
        float mu = s * (1.f / RD_);
        float var = q * (1.f / RD_) - mu * mu;
        sDegO[0] = mu;
        sDegO[1] = rsqrtf(var + EPS_);
    }
    __syncthreads();
    emb_out[(size_t)p * RD_ + tid] = lrelu((sR[384 + tid] - sDegO[0]) * sDegO[1]);
}

// ---------------------------------------------------------------------------
// Mesh-stage kernels
// ---------------------------------------------------------------------------
__global__ __launch_bounds__(256)
void zero_kernel(float* __restrict__ p, int n) {
    int i = blockIdx.x * blockDim.x + threadIdx.x;
    if (i < n) p[i] = 0.f;
}

__global__ __launch_bounds__(256)
void mesh_deg_kernel(const int* __restrict__ src, const int* __restrict__ dst,
                     float* __restrict__ degO, float* __restrict__ degI) {
    int e = blockIdx.x * blockDim.x + threadIdx.x;
    if (e < EM_) {
        atomicAdd(&degO[src[e]], 1.f);
        atomicAdd(&degI[dst[e]], 1.f);
    }
}

__global__ __launch_bounds__(256)
void deg_finalize_kernel(float* __restrict__ degO, float* __restrict__ degI) {
    int i = blockIdx.x * blockDim.x + threadIdx.x;
    if (i < P_) {
        degO[i] = rsqrtf(fmaxf(degO[i], 1.f));
        degI[i] = rsqrtf(fmaxf(degI[i], 1.f));
    }
}

__global__ __launch_bounds__(256)
void mesh_scatter_kernel(const float* __restrict__ X, int cshift, int cmask,
                         const int* __restrict__ src, const int* __restrict__ dst,
                         const float* __restrict__ ew, const float* __restrict__ degOrs,
                         float* __restrict__ agg, int total) {
    int i = blockIdx.x * blockDim.x + threadIdx.x;
    if (i < total) {
        int e = i >> cshift;
        int c = i & cmask;
        int s = src[e];
        float v = X[((size_t)s << cshift) + c] * degOrs[s] * ew[e];
        atomicAdd(&agg[((size_t)dst[e] << cshift) + c], v);
    }
}

// Y[P,N] = (A * rsqrt(deg_in)[row]) @ W ; block = 4 waves; grid (P/16, N/64)
__global__ __launch_bounds__(128)
void mesh_gemm_kernel(const float* __restrict__ A, int K, const float* __restrict__ rsI,
                      const float* __restrict__ W, int N, float* __restrict__ Y) {
    const int lane = threadIdx.x & 31;
    const int wave = threadIdx.x >> 5;
    const int row_base = blockIdx.x * 16;
    const int col_base = blockIdx.y * 64 + wave * 16;
    v8f acc = {};
    for (int k = 0; k < K; k += 32) {
        if (k + 32 < K) {
            __builtin_prefetch(&A[(size_t)row_base * K + k + 32], 0, 1);
            __builtin_prefetch(&W[(size_t)(k + 32) * N + col_base], 0, 1);
        }
        v16h a = load_a_f32_scaled(A, K, row_base, k, lane, rsI);
        v16h b = load_b_f32(W, N, k, col_base, lane);
        acc = wmma_f16(a, b, acc);
    }
    store_d_f32(Y, N, row_base, col_base, lane, acc);
}

// per-channel sum / sum-of-squares over rows; one block per channel
__global__ __launch_bounds__(256)
void colstat_kernel(const float* __restrict__ Y, int C, int nrows,
                    float* __restrict__ sumOut, float* __restrict__ sqOut) {
    __shared__ float ss[256], sq[256];
    const int c = blockIdx.x;
    float s = 0.f, q = 0.f;
    for (int r = threadIdx.x; r < nrows; r += 256) {
        float v = Y[(size_t)r * C + c];
        s += v; q += v * v;
    }
    ss[threadIdx.x] = s; sq[threadIdx.x] = q;
    __syncthreads();
    for (int off = 128; off > 0; off >>= 1) {
        if (threadIdx.x < off) {
            ss[threadIdx.x] += ss[threadIdx.x + off];
            sq[threadIdx.x] += sq[threadIdx.x + off];
        }
        __syncthreads();
    }
    if (threadIdx.x == 0) { sumOut[c] = ss[0]; sqOut[c] = sq[0]; }
}

__global__ __launch_bounds__(256)
void norm_apply_kernel(float* __restrict__ Y, int cmask,
                       const float* __restrict__ sum, const float* __restrict__ sq,
                       const float* __restrict__ g, const float* __restrict__ b,
                       int n, float invn) {
    int i = blockIdx.x * blockDim.x + threadIdx.x;
    if (i < n) {
        int c = i & cmask;
        float mu = sum[c] * invn;
        float var = sq[c] * invn - mu * mu;
        float v = g[c] * (Y[i] - mu) * rsqrtf(var + EPS_) + b[c];
        Y[i] = lrelu(v);
    }
}

__global__ __launch_bounds__(256)
void scale_copy_kernel(float* __restrict__ dstp, const float* __restrict__ srcp, int n, float s) {
    int i = blockIdx.x * blockDim.x + threadIdx.x;
    if (i < n) dstp[i] = srcp[i] * s;
}

__global__ __launch_bounds__(32)
void final_kernel(const float* __restrict__ ro, const float* __restrict__ Wcls, float* __restrict__ out) {
    int c = threadIdx.x;
    if (c < OUT_) {
        float acc = 0.f;
        for (int k = 0; k < 2 * HIDM_; ++k) acc += ro[k] * Wcls[k * OUT_ + c];
        out[c] = acc;
    }
}

// ---------------------------------------------------------------------------
extern "C" void kernel_launch(void* const* d_in, const int* in_sizes, int n_in,
                              void* d_out, int out_size, void* d_ws, size_t ws_size,
                              hipStream_t stream) {
    const float* patch_feats = (const float*)d_in[0];
    const int*   patch_src   = (const int*)d_in[1];
    const int*   patch_dst   = (const int*)d_in[2];
    const float* patch_ew    = (const float*)d_in[3];
    const int*   mesh_src    = (const int*)d_in[4];
    const int*   mesh_dst    = (const int*)d_in[5];
    const float* mesh_ew     = (const float*)d_in[6];
    const float* W1   = (const float*)d_in[7];
    const float* g1   = (const float*)d_in[8];
    const float* b1   = (const float*)d_in[9];
    const float* W2   = (const float*)d_in[10];
    const float* g2   = (const float*)d_in[11];
    const float* b2   = (const float*)d_in[12];
    const float* We   = (const float*)d_in[13];
    const float* Wc1  = (const float*)d_in[14];
    const float* g3   = (const float*)d_in[15];
    const float* b3   = (const float*)d_in[16];
    const float* Wc2  = (const float*)d_in[17];
    const float* g4   = (const float*)d_in[18];
    const float* b4   = (const float*)d_in[19];
    const float* Wcls = (const float*)d_in[20];
    float* out = (float*)d_out;

    // workspace layout (floats)
    float* ws    = (float*)d_ws;
    float* emb   = ws;                               // P * 128
    float* degO  = emb  + (size_t)P_ * RD_;          // P
    float* degI  = degO + P_;                        // P
    float* agg   = degI + P_;                        // P * 256
    float* hbuf  = agg  + (size_t)P_ * HIDM_;        // P * 256
    float* stats = hbuf + (size_t)P_ * HIDM_;        // 512
    float* ro    = stats + 512;                      // 512

    // ---- patch embedder (WMMA + async-LDS DMA) ----
    patch_embed_kernel<<<P_, 128, 0, stream>>>(patch_feats, patch_src, patch_dst, patch_ew,
                                               W1, g1, b1, W2, g2, b2, We, emb);

    // ---- mesh degrees ----
    zero_kernel<<<(2 * P_ + 255) / 256, 256, 0, stream>>>(degO, 2 * P_);
    mesh_deg_kernel<<<(EM_ + 255) / 256, 256, 0, stream>>>(mesh_src, mesh_dst, degO, degI);
    deg_finalize_kernel<<<(P_ + 255) / 256, 256, 0, stream>>>(degO, degI);

    const float invP = 1.f / (float)P_;

    // ---- mesh layer 1: emb[P,128] -> hbuf[P,256] ----
    zero_kernel<<<((P_ * RD_) + 255) / 256, 256, 0, stream>>>(agg, P_ * RD_);
    {
        int total = EM_ * RD_;
        mesh_scatter_kernel<<<(total + 255) / 256, 256, 0, stream>>>(
            emb, 7, RD_ - 1, mesh_src, mesh_dst, mesh_ew, degO, agg, total);
    }
    mesh_gemm_kernel<<<dim3(P_ / 16, HIDM_ / 64), 128, 0, stream>>>(agg, RD_, degI, Wc1, HIDM_, hbuf);
    colstat_kernel<<<HIDM_, 256, 0, stream>>>(hbuf, HIDM_, P_, stats, stats + 256);
    norm_apply_kernel<<<((P_ * HIDM_) + 255) / 256, 256, 0, stream>>>(
        hbuf, HIDM_ - 1, stats, stats + 256, g3, b3, P_ * HIDM_, invP);
    colstat_kernel<<<HIDM_, 256, 0, stream>>>(hbuf, HIDM_, P_, stats, stats + 256);
    scale_copy_kernel<<<1, 256, 0, stream>>>(ro, stats, HIDM_, invP);

    // ---- mesh layer 2: hbuf[P,256] -> hbuf[P,256] (agg is the GEMM input) ----
    zero_kernel<<<((P_ * HIDM_) + 255) / 256, 256, 0, stream>>>(agg, P_ * HIDM_);
    {
        int total = EM_ * HIDM_;
        mesh_scatter_kernel<<<(total + 255) / 256, 256, 0, stream>>>(
            hbuf, 8, HIDM_ - 1, mesh_src, mesh_dst, mesh_ew, degO, agg, total);
    }
    mesh_gemm_kernel<<<dim3(P_ / 16, HIDM_ / 64), 128, 0, stream>>>(agg, HIDM_, degI, Wc2, HIDM_, hbuf);
    colstat_kernel<<<HIDM_, 256, 0, stream>>>(hbuf, HIDM_, P_, stats, stats + 256);
    norm_apply_kernel<<<((P_ * HIDM_) + 255) / 256, 256, 0, stream>>>(
        hbuf, HIDM_ - 1, stats, stats + 256, g4, b4, P_ * HIDM_, invP);
    colstat_kernel<<<HIDM_, 256, 0, stream>>>(hbuf, HIDM_, P_, stats, stats + 256);
    scale_copy_kernel<<<1, 256, 0, stream>>>(ro + HIDM_, stats, HIDM_, invP);

    // ---- classifier ----
    final_kernel<<<1, 32, 0, stream>>>(ro, Wcls, out);

    (void)in_sizes; (void)n_in; (void)out_size; (void)ws_size;
}
